// GAT_21758304322293
// MI455X (gfx1250) — matrix operand order repaired and checked by reading, not compiled, not run on hardware
//
#include <hip/hip_runtime.h>
#include <hip/hip_bf16.h>

// ---------------------------------------------------------------------------
// GAT forward, fused flash-style softmax, all GEMMs via v_wmma_f32_16x16x32_f16,
// adj tiles staged into LDS by the Tensor Data Mover (double-buffered).
// ---------------------------------------------------------------------------

typedef __attribute__((ext_vector_type(16))) _Float16 v16h;
typedef __attribute__((ext_vector_type(8)))  _Float16 v8h;
typedef __attribute__((ext_vector_type(8)))  float    v8f;
typedef __attribute__((ext_vector_type(4)))  unsigned int u32x4;
typedef __attribute__((ext_vector_type(8)))  int      i32x8;
typedef __attribute__((ext_vector_type(4)))  int      i32x4;

#define NN    3072
#define FIN   512
#define FHID  64
#define NH    8
#define NC    16
#define ALPHA 0.2f
#define NEGBIG (-9e15f)

#if defined(__has_builtin)
#  if __has_builtin(__builtin_amdgcn_tensor_load_to_lds) && \
      __has_builtin(__builtin_amdgcn_s_wait_tensorcnt)
#    define HAVE_TDM 1
#  else
#    define HAVE_TDM 0
#  endif
#else
#  define HAVE_TDM 0
#endif

// ---- workspace layout (bytes) ----
static constexpr size_t OFF_WH   = 0;                                   // f32 [NH][NN][FHID]
static constexpr size_t OFF_WHHT = OFF_WH   + (size_t)NH*NN*FHID*4;     // f16 [NH][FHID][NN]
static constexpr size_t OFF_WHT  = OFF_WHHT + (size_t)NH*FHID*NN*2;     // f16 [NH][FHID][FIN]
static constexpr size_t OFF_WOT  = OFF_WHT  + (size_t)NH*FHID*FIN*2;    // f16 [NC][FIN]
static constexpr size_t OFF_SRC1 = OFF_WOT  + (size_t)NC*FIN*2;         // f32 [NH][NN]
static constexpr size_t OFF_DST1 = OFF_SRC1 + (size_t)NH*NN*4;
static constexpr size_t OFF_M1   = OFF_DST1 + (size_t)NH*NN*4;
static constexpr size_t OFF_L1   = OFF_M1   + (size_t)NH*NN*4;
static constexpr size_t OFF_X2   = OFF_L1   + (size_t)NH*NN*4;          // f32 [NN][NH*FHID]
static constexpr size_t OFF_WH2  = OFF_X2   + (size_t)NN*NH*FHID*4;     // f32 [NN][NC]
static constexpr size_t OFF_WH2T = OFF_WH2  + (size_t)NN*NC*4;          // f16 [NC][NN]
static constexpr size_t OFF_SRC2 = OFF_WH2T + (size_t)NC*NN*2;          // f32 [NN]
static constexpr size_t OFF_DST2 = OFF_SRC2 + (size_t)NN*4;
static constexpr size_t OFF_M2   = OFF_DST2 + (size_t)NN*4;
static constexpr size_t OFF_L2   = OFF_M2   + (size_t)NN*4;
static constexpr size_t OFF_HOUT = OFF_L2   + (size_t)NN*4;             // f32 [NN][NC]

__device__ __forceinline__ float lrelu(float x) { return x > 0.f ? x : ALPHA * x; }

// online-softmax merge of (m,l) pairs
__device__ __forceinline__ void smerge(float& m, float& l, float m2, float l2) {
    float M = fmaxf(m, m2);
    l = l * __expf(m - M) + l2 * __expf(m2 - M);
    m = M;
}

// two 8-runs of f16 -> v16h (B fragment: col = lane&15, K = (e<8?e:e+8) + 8*kh)
__device__ __forceinline__ v16h load16h(const _Float16* p) {
    v8h lo = *(const v8h*)p;
    v8h hi = *(const v8h*)(p + 16);
    v16h r;
#pragma unroll
    for (int t = 0; t < 8; ++t) { r[t] = lo[t]; r[t + 8] = hi[t]; }
    return r;
}

// A fragment from an f32 row: element e holds K = (e<8?e:e+8) + 8*kh
__device__ __forceinline__ v16h cvtAfrag(const float* rowp, int k0, int kh) {
    const float4 f0 = *(const float4*)(rowp + k0 + 8 * kh);
    const float4 f1 = *(const float4*)(rowp + k0 + 8 * kh + 4);
    const float4 f2 = *(const float4*)(rowp + k0 + 8 * kh + 16);
    const float4 f3 = *(const float4*)(rowp + k0 + 8 * kh + 20);
    v16h a;
    a[0]=(_Float16)f0.x; a[1]=(_Float16)f0.y; a[2]=(_Float16)f0.z; a[3]=(_Float16)f0.w;
    a[4]=(_Float16)f1.x; a[5]=(_Float16)f1.y; a[6]=(_Float16)f1.z; a[7]=(_Float16)f1.w;
    a[8]=(_Float16)f2.x; a[9]=(_Float16)f2.y; a[10]=(_Float16)f2.z; a[11]=(_Float16)f2.w;
    a[12]=(_Float16)f3.x; a[13]=(_Float16)f3.y; a[14]=(_Float16)f3.z; a[15]=(_Float16)f3.w;
    return a;
}

#define WMMA_F16(A, B, C) \
    __builtin_amdgcn_wmma_f32_16x16x32_f16(false, (A), false, (B), (short)0, (C), false, false)

#if HAVE_TDM
// Issue a TDM load of a [16 rows][32 ints] tile of adj (row stride NN ints)
// into LDS at ldsAddr.  D# per CDNA5 ISA §8: group0 = {count|flags, lds_addr,
// global_addr[31:0], global_addr[56:32]|type=2}, group1 = {data_size=4B,
// tensor_dim0/1 = NN, tile = 32x16, tensor_dim0_stride = NN}.  Groups 2/3
// (3D..5D dims) and the trailing group are unused -> zero.
// This toolchain declares the 6-arg builtin:
//   (u32x4 g0, i32x8 g1, i32x4, i32x4, i32x8, i32 cpol)
__device__ __forceinline__ void tdm_load_adj_tile(const int* gptr, unsigned ldsAddr) {
    unsigned long long ga = (unsigned long long)(size_t)gptr;
    u32x4 g0;
    g0[0] = 1u;                                              // count=1 (valid), user mode
    g0[1] = ldsAddr;                                         // lds_addr (bytes)
    g0[2] = (unsigned)ga;                                    // global_addr[31:0]
    g0[3] = (unsigned)((ga >> 32) & 0x01FFFFFFu) | (2u << 30); // addr[56:32] | type=2
    i32x8 g1;
    g1[0] = (2 << 16);                                       // data_size=2 (4 bytes)
    g1[1] = (NN & 0xFFFF) << 16;                             // tensor_dim0 lo16
    g1[2] = ((NN >> 16) & 0xFFFF) | ((NN & 0xFFFF) << 16);   // dim0 hi16 | dim1 lo16
    g1[3] = ((NN >> 16) & 0xFFFF) | (32 << 16);              // dim1 hi16 | tile_dim0=32
    g1[4] = 16;                                              // tile_dim1=16, tile_dim2=0
    g1[5] = NN;                                              // tensor_dim0_stride lo32
    g1[6] = 0;                                               // stride0 hi | stride1 lo
    g1[7] = 0;
    i32x4 g2 = {0, 0, 0, 0};
    i32x4 g3 = {0, 0, 0, 0};
    i32x8 g4 = {0, 0, 0, 0, 0, 0, 0, 0};
    __builtin_amdgcn_tensor_load_to_lds(g0, g1, g2, g3, g4, 0);
}
#endif

// ---------------------------------------------------------------------------
// 0) weight transposes to f16 (B-fragment friendly layouts)
// ---------------------------------------------------------------------------
__global__ void k_prep(const float* __restrict__ Wheads, const float* __restrict__ Wout,
                       _Float16* __restrict__ WhT, _Float16* __restrict__ WoT) {
    int idx = blockIdx.x * blockDim.x + threadIdx.x;
    if (idx < NH * FHID * FIN) {
        int h = idx / (FHID * FIN), r = idx % (FHID * FIN);
        int o = r / FIN, f = r % FIN;
        WhT[idx] = (_Float16)Wheads[((size_t)h * FIN + f) * FHID + o];
    } else if (idx < NH * FHID * FIN + NC * FIN) {
        int r = idx - NH * FHID * FIN;
        int c = r / FIN, f = r % FIN;
        WoT[r] = (_Float16)Wout[(size_t)f * NC + c];
    }
}

// ---------------------------------------------------------------------------
// 1) Wh[h] = x @ W_heads[h].  wave = (head, 16-row tile); the A fragment is
//    converted once and reused across 4 column tiles (4 accumulators).
// ---------------------------------------------------------------------------
__global__ void __launch_bounds__(256) k_wh(const float* __restrict__ x,
                                            const _Float16* __restrict__ WhT,
                                            float* __restrict__ Wh,
                                            _Float16* __restrict__ WhhT) {
    int wv = blockIdx.x * (blockDim.x >> 5) + (threadIdx.x >> 5);   // 0..1535
    int h = wv / 192, it = wv % 192;
    int lane = threadIdx.x & 31, row = lane & 15, kh = lane >> 4;
    const float*    xrow = x   + (size_t)(it * 16 + row) * FIN;
    const _Float16* bp   = WhT + (size_t)h * FHID * FIN;
    v8f acc[4] = {};
#pragma unroll 2
    for (int k0 = 0; k0 < FIN; k0 += 32) {
        v16h a = cvtAfrag(xrow, k0, kh);
#pragma unroll
        for (int ob = 0; ob < 4; ++ob) {
            v16h b = load16h(bp + (size_t)(ob * 16 + row) * FIN + k0 + 8 * kh);
            acc[ob] = WMMA_F16(a, b, acc[ob]);
        }
    }
#pragma unroll
    for (int ob = 0; ob < 4; ++ob) {
        int o = ob * 16 + row;                 // C layout: N = lane&15
#pragma unroll
        for (int rr = 0; rr < 8; ++rr) {       // C layout: M = rr + 8*kh
            int n = it * 16 + rr + 8 * kh;
            float v = acc[ob][rr];
            Wh[((size_t)h * NN + n) * FHID + o] = v;
            WhhT[((size_t)h * FHID + o) * NN + n] = (_Float16)v;
        }
    }
}

// ---------------------------------------------------------------------------
// 2) src/dst projections, layer 1
// ---------------------------------------------------------------------------
__global__ void k_srcdst1(const float* __restrict__ Wh, const float* __restrict__ a_src,
                          const float* __restrict__ a_dst,
                          float* __restrict__ src1, float* __restrict__ dst1) {
    int idx = blockIdx.x * blockDim.x + threadIdx.x;   // NH*NN
    int h = idx / NN, n = idx % NN;
    const float* w  = Wh + ((size_t)h * NN + n) * FHID;
    const float* as = a_src + h * FHID;
    const float* ad = a_dst + h * FHID;
    float s = 0.f, d = 0.f;
#pragma unroll 8
    for (int o = 0; o < FHID; ++o) { float v = w[o]; s += v * as[o]; d += v * ad[o]; }
    src1[idx] = s; dst1[idx] = d;
}

// ---------------------------------------------------------------------------
// 3) row softmax stats (m,l), shared by both layers: block = one (h,i) row
// ---------------------------------------------------------------------------
__global__ void __launch_bounds__(256) k_stats(const int* __restrict__ adj,
                                               const float* __restrict__ src,
                                               const float* __restrict__ dst,
                                               float* __restrict__ mOut, float* __restrict__ lOut) {
    int b = blockIdx.x;
    int h = b / NN, i = b % NN;
    const int*   arow = adj + (size_t)i * NN;
    const float* drow = dst + (size_t)h * NN;
    float si = src[b];
    float m = -INFINITY, l = 0.f;
    for (int j = threadIdx.x; j < NN; j += 256) {
        float e = arow[j] > 0 ? lrelu(si + drow[j]) : NEGBIG;
        if (e > m) { l = l * __expf(m - e) + 1.f; m = e; }
        else        l += __expf(e - m);
    }
#pragma unroll
    for (int off = 16; off; off >>= 1) {
        float m2 = __shfl_xor(m, off, 32);
        float l2 = __shfl_xor(l, off, 32);
        smerge(m, l, m2, l2);
    }
    __shared__ float sm[8], sl[8];
    if ((threadIdx.x & 31) == 0) { sm[threadIdx.x >> 5] = m; sl[threadIdx.x >> 5] = l; }
    __syncthreads();
    if (threadIdx.x == 0) {
        for (int w = 1; w < 8; ++w) smerge(m, l, sm[w], sl[w]);
        mOut[b] = m; lOut[b] = l;
    }
}

// ---------------------------------------------------------------------------
// 4) h' = att @ Wh, fused ELU + concat -> x2.  block = 16-row tile, wave = head.
//    adj tile staged into LDS by the Tensor Data Mover, double-buffered with
//    TENSORcnt; attention probs regenerated per lane into the A fragment.
// ---------------------------------------------------------------------------
__global__ void __launch_bounds__(256) k_attgemm(const int* __restrict__ adj,
                                                 const float* __restrict__ src1,
                                                 const float* __restrict__ dst1,
                                                 const float* __restrict__ m1,
                                                 const float* __restrict__ l1,
                                                 const _Float16* __restrict__ WhhT,
                                                 float* __restrict__ x2) {
#if HAVE_TDM
    __shared__ int adjT[2][16 * 32];
#else
    __shared__ int adjT[1][16 * 32];
#endif
    __shared__ float dstS[NH * 32];
    int h = threadIdx.x >> 5;
    int lane = threadIdx.x & 31, row = lane & 15, kh = lane >> 4;
    int i = blockIdx.x * 16 + row;
    float si   = src1[h * NN + i];
    float mi   = m1[h * NN + i];
    float linv = 1.f / l1[h * NN + i];
    v8f acc[4] = {};
    const _Float16* bbase = WhhT + (size_t)h * FHID * NN;
    const int* tilebase = adj + (size_t)(blockIdx.x * 16) * NN;

#if HAVE_TDM
    if (h == 0)   // one wave issues the DMA; TDM ignores EXEC, issues once per wave
        tdm_load_adj_tile(tilebase, (unsigned)(size_t)&adjT[0][0]);
#endif

    for (int j0 = 0; j0 < NN; j0 += 32) {
#if HAVE_TDM
        const int buf = (j0 >> 5) & 1;
#else
        const int buf = 0;
#endif
        __syncthreads();   // everyone done reading the buffer the next DMA overwrites
        {
            int t = threadIdx.x;
            dstS[t] = dst1[(size_t)(t >> 5) * NN + j0 + (t & 31)];
#if !HAVE_TDM
            adjT[0][t]       = tilebase[(size_t)(t >> 5) * NN + j0 + (t & 31)];
            adjT[0][t + 256] = tilebase[(size_t)((t + 256) >> 5) * NN + j0 + (t & 31)];
            if (j0 + 32 < NN)
                __builtin_prefetch(tilebase + (size_t)(t >> 5) * NN + j0 + 32 + (t & 31), 0, 1);
#endif
        }
#if HAVE_TDM
        if (h == 0) {
            if (j0 + 32 < NN) {   // prefetch next tile, then wait for current one
                tdm_load_adj_tile(tilebase + j0 + 32, (unsigned)(size_t)&adjT[buf ^ 1][0]);
                __builtin_amdgcn_s_wait_tensorcnt(1);
            } else {
                __builtin_amdgcn_s_wait_tensorcnt(0);
            }
        }
#endif
        __syncthreads();   // current tile visible to all 8 head-waves
        v16h a;
#pragma unroll
        for (int e = 0; e < 16; ++e) {
            int K = (e < 8 ? e : e + 8) + 8 * kh;    // 16-bit A-fragment K map
            float ev = adjT[buf][row * 32 + K] > 0 ? lrelu(si + dstS[h * 32 + K]) : NEGBIG;
            a[e] = (_Float16)(__expf(ev - mi) * linv);
        }
#pragma unroll
        for (int ob = 0; ob < 4; ++ob) {
            v16h b = load16h(bbase + (size_t)(ob * 16 + row) * NN + j0 + 8 * kh);
            acc[ob] = WMMA_F16(a, b, acc[ob]);
        }
    }
#pragma unroll
    for (int ob = 0; ob < 4; ++ob) {
#pragma unroll
        for (int rr = 0; rr < 8; ++rr) {
            int n = blockIdx.x * 16 + rr + 8 * kh;
            float v = acc[ob][rr];
            x2[(size_t)n * (NH * FHID) + h * FHID + ob * 16 + row] =
                v > 0.f ? v : __expf(v) - 1.f;   // ELU
        }
    }
}

// ---------------------------------------------------------------------------
// 5) avg_attention -> d_out (one thread per (i,j), loops heads)
// ---------------------------------------------------------------------------
__global__ void k_avg(const int* __restrict__ adj, const float* __restrict__ src1,
                      const float* __restrict__ dst1, const float* __restrict__ m1,
                      const float* __restrict__ l1, float* __restrict__ avg) {
    int b = blockIdx.x;
    int i = b / (NN / 256);
    int j = (b % (NN / 256)) * 256 + threadIdx.x;
    int a = adj[(size_t)i * NN + j];
    float s = 0.f;
#pragma unroll
    for (int h = 0; h < NH; ++h) {
        float e = a > 0 ? lrelu(src1[h * NN + i] + dst1[h * NN + j]) : NEGBIG;
        s += __expf(e - m1[h * NN + i]) * (1.f / l1[h * NN + i]);
    }
    avg[(size_t)i * NN + j] = s * (1.f / NH);
}

// ---------------------------------------------------------------------------
// 6) Wh2 = x2 @ W_out  (wave = 16-row tile, single 16-col tile)
// ---------------------------------------------------------------------------
__global__ void __launch_bounds__(256) k_wh2(const float* __restrict__ x2,
                                             const _Float16* __restrict__ WoT,
                                             float* __restrict__ Wh2,
                                             _Float16* __restrict__ Wh2T) {
    int wv = blockIdx.x * (blockDim.x >> 5) + (threadIdx.x >> 5);   // 0..191
    int lane = threadIdx.x & 31, row = lane & 15, kh = lane >> 4;
    const float*    xrow = x2  + (size_t)(wv * 16 + row) * FIN;
    const _Float16* bp   = WoT + (size_t)row * FIN;
    v8f acc = {};
#pragma unroll 4
    for (int k0 = 0; k0 < FIN; k0 += 32) {
        v16h a = cvtAfrag(xrow, k0, kh);
        v16h b = load16h(bp + k0 + 8 * kh);
        acc = WMMA_F16(a, b, acc);
    }
#pragma unroll
    for (int rr = 0; rr < 8; ++rr) {
        int n = wv * 16 + rr + 8 * kh;
        float v = acc[rr];
        Wh2[(size_t)n * NC + row] = v;
        Wh2T[(size_t)row * NN + n] = (_Float16)v;
    }
}

// ---------------------------------------------------------------------------
// 7) src2/dst2 projections
// ---------------------------------------------------------------------------
__global__ void k_srcdst2(const float* __restrict__ Wh2, const float* __restrict__ aos,
                          const float* __restrict__ aod,
                          float* __restrict__ src2, float* __restrict__ dst2) {
    int n = blockIdx.x * blockDim.x + threadIdx.x;
    const float* w = Wh2 + (size_t)n * NC;
    float s = 0.f, d = 0.f;
#pragma unroll
    for (int c = 0; c < NC; ++c) { s += w[c] * aos[c]; d += w[c] * aod[c]; }
    src2[n] = s; dst2[n] = d;
}

// ---------------------------------------------------------------------------
// 8) h_out = att2 @ Wh2  (wave = 16-row tile; adj/dst2 read as b128 runs)
// ---------------------------------------------------------------------------
__global__ void __launch_bounds__(256) k_outgemm(const int* __restrict__ adj,
                                                 const float* __restrict__ src2,
                                                 const float* __restrict__ dst2,
                                                 const float* __restrict__ m2,
                                                 const float* __restrict__ l2,
                                                 const _Float16* __restrict__ Wh2T,
                                                 float* __restrict__ hout) {
    int wv = blockIdx.x * (blockDim.x >> 5) + (threadIdx.x >> 5);   // 0..191
    int lane = threadIdx.x & 31, row = lane & 15, kh = lane >> 4;
    int i = wv * 16 + row;
    const int* arow = adj + (size_t)i * NN;
    float si = src2[i], mi = m2[i], linv = 1.f / l2[i];
    const _Float16* bp = Wh2T + (size_t)row * NN;
    v8f acc = {};
    for (int j0 = 0; j0 < NN; j0 += 32) {
        const int base0 = j0 + 8 * kh;
        int4   A0 = *(const int4*)(arow + base0);
        int4   A1 = *(const int4*)(arow + base0 + 4);
        int4   A2 = *(const int4*)(arow + base0 + 16);
        int4   A3 = *(const int4*)(arow + base0 + 20);
        float4 D0 = *(const float4*)(dst2 + base0);
        float4 D1 = *(const float4*)(dst2 + base0 + 4);
        float4 D2 = *(const float4*)(dst2 + base0 + 16);
        float4 D3 = *(const float4*)(dst2 + base0 + 20);
        if (j0 + 32 < NN) __builtin_prefetch(arow + j0 + 32 + 8 * kh, 0, 1);
        int   av[16] = {A0.x,A0.y,A0.z,A0.w, A1.x,A1.y,A1.z,A1.w,
                        A2.x,A2.y,A2.z,A2.w, A3.x,A3.y,A3.z,A3.w};
        float dv[16] = {D0.x,D0.y,D0.z,D0.w, D1.x,D1.y,D1.z,D1.w,
                        D2.x,D2.y,D2.z,D2.w, D3.x,D3.y,D3.z,D3.w};
        v16h a;
#pragma unroll
        for (int e = 0; e < 16; ++e) {
            float ev = av[e] > 0 ? lrelu(si + dv[e]) : NEGBIG;
            a[e] = (_Float16)(__expf(ev - mi) * linv);
        }
        v16h b = load16h(bp + j0 + 8 * kh);
        acc = WMMA_F16(a, b, acc);
    }
#pragma unroll
    for (int rr = 0; rr < 8; ++rr)
        hout[(size_t)(wv * 16 + rr + 8 * kh) * NC + row] = acc[rr];
}

// ---------------------------------------------------------------------------
// 9) log_softmax over 16 classes -> d_out
// ---------------------------------------------------------------------------
__global__ void k_logsm(const float* __restrict__ hout, float* __restrict__ logp) {
    int n = blockIdx.x * blockDim.x + threadIdx.x;
    const float* r = hout + (size_t)n * NC;
    float mx = r[0];
#pragma unroll
    for (int c = 1; c < NC; ++c) mx = fmaxf(mx, r[c]);
    float s = 0.f;
#pragma unroll
    for (int c = 0; c < NC; ++c) s += __expf(r[c] - mx);
    float lse = mx + __logf(s);
#pragma unroll
    for (int c = 0; c < NC; ++c) logp[(size_t)n * NC + c] = r[c] - lse;
}

// ---------------------------------------------------------------------------
extern "C" void kernel_launch(void* const* d_in, const int* in_sizes, int n_in,
                              void* d_out, int out_size, void* d_ws, size_t ws_size,
                              hipStream_t stream) {
    (void)in_sizes; (void)n_in; (void)out_size; (void)ws_size;
    const float* x      = (const float*)d_in[0];
    const int*   adj    = (const int*)d_in[1];
    const float* Wheads = (const float*)d_in[2];
    const float* a_src  = (const float*)d_in[3];
    const float* a_dst  = (const float*)d_in[4];
    const float* Wout   = (const float*)d_in[5];
    const float* aos    = (const float*)d_in[6];
    const float* aod    = (const float*)d_in[7];
    float* out = (float*)d_out;                    // [NN*NC logp][NN*NN avg]
    char*  ws  = (char*)d_ws;

    float*    Wh    = (float*)(ws + OFF_WH);
    _Float16* WhhT  = (_Float16*)(ws + OFF_WHHT);
    _Float16* WhT   = (_Float16*)(ws + OFF_WHT);
    _Float16* WoT   = (_Float16*)(ws + OFF_WOT);
    float*    src1  = (float*)(ws + OFF_SRC1);
    float*    dst1  = (float*)(ws + OFF_DST1);
    float*    m1    = (float*)(ws + OFF_M1);
    float*    l1    = (float*)(ws + OFF_L1);
    float*    x2    = (float*)(ws + OFF_X2);
    float*    Wh2   = (float*)(ws + OFF_WH2);
    _Float16* Wh2T  = (_Float16*)(ws + OFF_WH2T);
    float*    src2  = (float*)(ws + OFF_SRC2);
    float*    dst2  = (float*)(ws + OFF_DST2);
    float*    m2    = (float*)(ws + OFF_M2);
    float*    l2    = (float*)(ws + OFF_L2);
    float*    hout  = (float*)(ws + OFF_HOUT);

    k_prep   <<<(NH*FHID*FIN + NC*FIN + 255) / 256, 256, 0, stream>>>(Wheads, Wout, WhT, WoT);
    k_wh     <<<192,        256, 0, stream>>>(x, WhT, Wh, WhhT);
    k_srcdst1<<<NH*NN/256,  256, 0, stream>>>(Wh, a_src, a_dst, src1, dst1);
    k_stats  <<<NH*NN,      256, 0, stream>>>(adj, src1, dst1, m1, l1);
    k_attgemm<<<NN/16,      256, 0, stream>>>(adj, src1, dst1, m1, l1, WhhT, x2);
    k_avg    <<<NN*(NN/256),256, 0, stream>>>(adj, src1, dst1, m1, l1, out + (size_t)NN*NC);
    k_wh2    <<<24,         256, 0, stream>>>(x2, WoT, Wh2, Wh2T);
    k_srcdst2<<<NN/256,     256, 0, stream>>>(Wh2, aos, aod, src2, dst2);
    k_stats  <<<NN,         256, 0, stream>>>(adj, src2, dst2, m2, l2);
    k_outgemm<<<24,         256, 0, stream>>>(adj, src2, dst2, m2, l2, Wh2T, hout);
    k_logsm  <<<NN/256,     256, 0, stream>>>(hout, out);
}
